// Model_23046794510595
// MI455X (gfx1250) — matrix operand order repaired and checked by reading, not compiled, not run on hardware
//
#include <hip/hip_runtime.h>
#include <stdint.h>

// ---------------------------------------------------------------------------
// MI455X (gfx1250) 2-layer LSTM + linear head.
//   Phase 1: xg = input @ W_ih^T + (b_ih+b_hh)   (full-chip bf16 WMMA GEMM)
//   Phase 2: persistent recurrence kernel, W_hh slice LDS-resident,
//            16 batch-groups x 16 WGs, cluster-barrier + atomic group sync.
//   Phase 3: FC head via one WMMA tile per block.
// Requires ws_size >= ~1.22 GB (layout below).
// ---------------------------------------------------------------------------

typedef unsigned short u16;
typedef unsigned int   u32;
typedef unsigned long long u64;

typedef __attribute__((ext_vector_type(16))) __bf16 bf16x16;
typedef __attribute__((ext_vector_type(8)))  float  f32x8;
typedef __attribute__((ext_vector_type(4)))  unsigned int uintx4;
typedef __attribute__((ext_vector_type(8)))  int    intx8;
typedef __attribute__((ext_vector_type(4)))  int    intx4;

#define HIDDEN 512
#define GATES  2048   // 4*HIDDEN, PyTorch order i,f,g,o
#define NKEYS  128
#define BATCH  256
#define SEQ    512
#define MROWS  (BATCH * SEQ)   // 131072

// ---- workspace layout (bytes, all 256-aligned) ----
#define BAR_OFF    0ull          //  64 u32: ctr[16], gen[16], pad
#define BIAS0_OFF  256ull        //  2048 f32
#define BIAS1_OFF  8448ull       //  2048 f32
#define WIH0_OFF   16640ull      //  2048x128  bf16
#define WHH0_OFF   540928ull     //  2048x512  bf16
#define WIH1_OFF   2638080ull    //  2048x512  bf16
#define WHH1_OFF   4735232ull    //  2048x512  bf16
#define WFC_OFF    6832384ull    //  128x512   bf16
#define HST_OFF    6963456ull    //  256x512   bf16  (current h, also h_last)
#define HSEQ_OFF   7225600ull    //  256x512x512 bf16 (layer-0 output seq)
#define XG_OFF     141443328ull  //  131072x2048 f32 (reused by both layers)
// total required: 1215185152 bytes

#define LSTM_SMEM 157696  // 128KB Whh slice + 16KB h + 8KB gates + 2KB c

#ifndef USE_TDM
#define USE_TDM 1
#endif
#if USE_TDM && defined(__has_builtin)
#  if __has_builtin(__builtin_amdgcn_tensor_load_to_lds)
#    define TDM_OK 1
#  endif
#endif
#ifndef TDM_OK
#  define TDM_OK 0
#endif

// ---------------------------------------------------------------------------
// helpers
// ---------------------------------------------------------------------------
__device__ __forceinline__ u16 f2bf(float f) {            // RNE f32 -> bf16
  u32 u = __builtin_bit_cast(u32, f);
  return (u16)((u + 0x7fffu + ((u >> 16) & 1u)) >> 16);
}
__device__ __forceinline__ float sigmoidf_(float x) { return 1.0f / (1.0f + __expf(-x)); }

union BF16Frag { u32 u[8]; bf16x16 v; };

__device__ __forceinline__ f32x8 wmma_bf16(bf16x16 a, bf16x16 b, f32x8 c) {
  // emits v_wmma_f32_16x16x32_bf16
  return __builtin_amdgcn_wmma_f32_16x16x32_bf16(false, a, false, b, (short)0, c, false, false);
}

// A fragment (16x32 bf16, MxK): lane holds row M=l16; VGPR r holds
// K = (r>=4?16:0) + half*8 + (r&3)*2 (+1).  p points at 16-row block, stride ld.
__device__ __forceinline__ bf16x16 load_a(const u16* p, int ld, int kBase, int l16, int half) {
  BF16Frag F;
#pragma unroll
  for (int r = 0; r < 8; ++r) {
    int kk = kBase + ((r & 4) ? 16 : 0) + half * 8 + (r & 3) * 2;
    F.u[r] = *(const u32*)(p + (size_t)l16 * ld + kk);
  }
  return F.v;
}
// B fragment (32x16 bf16, KxN) from row-major W[n][k]: lane holds col N=l16;
// VGPR r holds K = (r>=4?16:0) + half*8 + (r&3)*2 (+1).  p points at 16-col (row) block, stride ld.
__device__ __forceinline__ bf16x16 load_b(const u16* p, int ld, int kBase, int l16, int half) {
  BF16Frag F;
#pragma unroll
  for (int r = 0; r < 8; ++r) {
    int kk = kBase + ((r & 4) ? 16 : 0) + half * 8 + (r & 3) * 2;
    F.u[r] = *(const u32*)(p + (size_t)l16 * ld + kk);
  }
  return F.v;
}

// ---------------------------------------------------------------------------
// tiny prep kernels
// ---------------------------------------------------------------------------
__global__ void k_f2b(u16* dst, const float* src, int n) {
  int i = blockIdx.x * 256 + threadIdx.x;
  if (i < n) dst[i] = f2bf(src[i]);
}
__global__ void k_bias(float* dst, const float* a, const float* b, int n) {
  int i = blockIdx.x * 256 + threadIdx.x;
  if (i < n) dst[i] = a[i] + b[i];
}
__global__ void k_zero(u32* p, int n) {
  int i = blockIdx.x * 256 + threadIdx.x;
  if (i < n) p[i] = 0u;
}

// ---------------------------------------------------------------------------
// Phase 1: xg[m][g] = A[m][:] . W[g][:] + bias[g]   (N fixed = 2048)
// block: 256 thr (8 waves), tile 128x128, whole K staged in LDS once.
// ---------------------------------------------------------------------------
template<int K, bool AF32>
__global__ void __launch_bounds__(256) k_xg_gemm(float* __restrict__ xg,
    const void* __restrict__ Asrc, const u16* __restrict__ W,
    const float* __restrict__ bias)
{
  extern __shared__ char smem[];
  u16* As = (u16*)smem;                   // [128][K]
  u16* Bs = (u16*)(smem + 128 * K * 2);   // [128][K]
  const int nb = blockIdx.x, mb = blockIdx.y;
  const int mbase = mb * 128, nbase = nb * 128;
  const int tid = threadIdx.x;

  if (AF32) {
    const float* A = (const float*)Asrc + (size_t)mbase * K;
    for (int i = tid; i < 128 * K; i += 256) As[i] = f2bf(A[i]);
  } else {
    const uint4* A = (const uint4*)((const u16*)Asrc + (size_t)mbase * K);
    uint4* D = (uint4*)As;
    for (int i = tid; i < 128 * K / 8; i += 256) D[i] = A[i];
  }
  {
    const uint4* S = (const uint4*)(W + (size_t)nbase * K);
    uint4* D = (uint4*)Bs;
    for (int i = tid; i < 128 * K / 8; i += 256) D[i] = S[i];
  }
  __syncthreads();

  const int wave = tid >> 5, lane = tid & 31, half = lane >> 4, l16 = lane & 15;
  f32x8 acc[8];
#pragma unroll
  for (int n = 0; n < 8; ++n) {
    float bv = bias[nbase + n * 16 + l16];
#pragma unroll
    for (int v = 0; v < 8; ++v) acc[n][v] = bv;
  }
  const u16* Arow = As + (size_t)(wave * 16) * K;
#pragma unroll 4
  for (int kt = 0; kt < K / 32; ++kt) {
    bf16x16 a = load_a(Arow, K, kt * 32, l16, half);
#pragma unroll
    for (int n = 0; n < 8; ++n) {
      bf16x16 b = load_b(Bs + (size_t)(n * 16) * K, K, kt * 32, l16, half);
      acc[n] = wmma_bf16(a, b, acc[n]);
    }
  }
#pragma unroll
  for (int n = 0; n < 8; ++n)
#pragma unroll
    for (int v = 0; v < 8; ++v)
      xg[(size_t)(mbase + wave * 16 + v + half * 8) * GATES + (nbase + n * 16 + l16)] = acc[n][v];
}

// ---------------------------------------------------------------------------
// group barrier: 16 WGs of a batch-group. Cluster barrier (CDNA5, NOP when not
// cluster-dispatched) + authoritative atomic generation barrier in L2.
// ---------------------------------------------------------------------------
__device__ void group_barrier(u32* ctr, u32* gen) {
  __syncthreads();
  __threadfence();
  if (threadIdx.x == 0) {
    __builtin_amdgcn_s_cluster_barrier();   // s_barrier_signal/wait -3
    u32 g = __hip_atomic_load(gen, __ATOMIC_ACQUIRE, __HIP_MEMORY_SCOPE_AGENT);
    u32 a = __hip_atomic_fetch_add(ctr, 1u, __ATOMIC_ACQ_REL, __HIP_MEMORY_SCOPE_AGENT);
    if (a == 15u) {
      __hip_atomic_store(ctr, 0u, __ATOMIC_RELAXED, __HIP_MEMORY_SCOPE_AGENT);
      __hip_atomic_fetch_add(gen, 1u, __ATOMIC_ACQ_REL, __HIP_MEMORY_SCOPE_AGENT);
    } else {
      while (__hip_atomic_load(gen, __ATOMIC_ACQUIRE, __HIP_MEMORY_SCOPE_AGENT) == g)
        __builtin_amdgcn_s_sleep(2);
    }
  }
  __syncthreads();
  __threadfence();
}

// ---------------------------------------------------------------------------
// Phase 2: persistent recurrence. grid = 256 WGs (group = wg>>4 owns 16 batch
// rows; nidx = wg&15 owns 32 hidden units => 4x32 gate rows of W_hh in LDS).
// wave w: gate q=w>>1, N-half=w&1 -> one 16x16 WMMA tile, K=512 over LDS h.
// ---------------------------------------------------------------------------
__global__ void __launch_bounds__(256) k_lstm(const float* __restrict__ xg,
    const u16* __restrict__ Whh, u16* __restrict__ hstate,
    u16* __restrict__ hseq, u32* __restrict__ bar, int writeSeq)
{
  extern __shared__ char smem[];
  u16*   Wl = (u16*)smem;                        // [4*32][512] gate-row slice
  u16*   u  = (u16*)(smem + 131072);             // [16][512] h_{t-1} (bf16)
  float* gb = (float*)(smem + 131072 + 16384);   // [4][16][32] raw gates
  float* cb = gb + 4 * 16 * 32;                  // [16][32] cell state (f32)

  const int wg = blockIdx.x, group = wg >> 4, nidx = wg & 15;
  const int bbase = group * 16, hidbase = nidx * 32;
  const int tid = threadIdx.x, wave = tid >> 5, lane = tid & 31;
  const int half = lane >> 4, l16 = lane & 15;
  const int q = wave >> 1, halfn = wave & 1;

#if TDM_OK
  // TDM: DMA each gate's 32x512 bf16 row-block (contiguous) into LDS.
  if (wave < 4) {
    const int qq = wave;
    u32 lds_off = (u32)(u64)(uintptr_t)(Wl + (size_t)qq * 32 * HIDDEN); // LDS aperture low bits
    u64 ga = (u64)(uintptr_t)(Whh + ((size_t)(qq * HIDDEN) + hidbase) * HIDDEN);
    uintx4 g0;
    g0[0] = 1u;                                        // count=1, user descriptor
    g0[1] = lds_off;                                   // lds_addr
    g0[2] = (u32)(ga & 0xffffffffu);                   // global_addr[31:0]
    g0[3] = (u32)((ga >> 32) & 0x1ffffffu) | (2u << 30); // addr[56:32] | type=2
    intx8 g1;
    g1[0] = (int)(1u << 16);                 // data_size=1 (2 bytes/elem)
    g1[1] = (int)((u32)HIDDEN << 16);        // tensor_dim0 = 512 (bits 63:48)
    g1[2] = (int)(32u << 16);                // tensor_dim1 = 32  (bits 95:80)
    g1[3] = (int)((u32)HIDDEN << 16);        // tile_dim0   = 512 (bits 127:112)
    g1[4] = 32;                              // tile_dim1 = 32, tile_dim2 = 0
    g1[5] = HIDDEN;                          // tensor_dim0_stride = 512
    g1[6] = 0; g1[7] = 0;
    intx4 g2 = (intx4)0, g3 = (intx4)0;      // groups 2/3 unused (2-D tile)
    intx8 g4 = (intx8)0;                     // trailing group (6-arg toolchain)
    __builtin_amdgcn_tensor_load_to_lds(g0, g1, g2, g3, g4, 0);
    __builtin_amdgcn_s_wait_tensorcnt(0);
  }
#endif
  // authoritative cooperative copy of the same slice (covers TDM semantics risk)
#pragma unroll
  for (int qq = 0; qq < 4; ++qq) {
    const uint4* S = (const uint4*)(Whh + ((size_t)(qq * HIDDEN) + hidbase) * HIDDEN);
    uint4* D = (uint4*)(Wl + (size_t)qq * 32 * HIDDEN);
    for (int i = tid; i < 32 * HIDDEN / 8; i += 256) D[i] = S[i];
  }
  for (int i = tid; i < 16 * 32; i += 256) cb[i] = 0.0f;
  {
    uint4 z; z.x = z.y = z.z = z.w = 0u;
    uint4* U4 = (uint4*)u;
    for (int i = tid; i < 16 * HIDDEN / 8; i += 256) U4[i] = z;  // h_{-1} = 0
  }
  __syncthreads();

  const int colGlobal = q * HIDDEN + hidbase + halfn * 16 + l16;
  u32* ctr = bar + group;
  u32* gen = bar + 16 + group;
  const u16* Brow = Wl + (size_t)(q * 32 + halfn * 16) * HIDDEN;

  for (int t = 0; t < SEQ; ++t) {
    // init accumulator from precomputed input contribution (bias folded in)
    f32x8 acc;
#pragma unroll
    for (int v = 0; v < 8; ++v) {
      int m = v + half * 8;
      acc[v] = xg[((size_t)(bbase + m) * SEQ + t) * GATES + colGlobal];
    }
    if (t + 1 < SEQ)  // warm L2/WGP$ for next step's gate slice
      __builtin_prefetch((const void*)&xg[((size_t)(bbase + l16) * SEQ + (t + 1)) * GATES + colGlobal], 0, 1);

#pragma unroll
    for (int kt = 0; kt < HIDDEN / 32; ++kt) {
      bf16x16 a = load_a(u, HIDDEN, kt * 32, l16, half);
      bf16x16 b = load_b(Brow, HIDDEN, kt * 32, l16, half);
      acc = wmma_bf16(a, b, acc);
    }
#pragma unroll
    for (int v = 0; v < 8; ++v)
      gb[(q * 16 + v + half * 8) * 32 + halfn * 16 + l16] = acc[v];
    __syncthreads();

    // fused gate nonlinearities + cell/h update for our 16x32 slice
    for (int e = tid; e < 16 * 32; e += 256) {
      int m = e >> 5, j = e & 31;
      float gi = gb[(0 * 16 + m) * 32 + j];
      float gf = gb[(1 * 16 + m) * 32 + j];
      float gg = gb[(2 * 16 + m) * 32 + j];
      float go = gb[(3 * 16 + m) * 32 + j];
      float c = sigmoidf_(gf) * cb[e] + sigmoidf_(gi) * tanhf(gg);
      cb[e] = c;
      float h = sigmoidf_(go) * tanhf(c);
      u16 hb = f2bf(h);
      hstate[(size_t)(bbase + m) * HIDDEN + hidbase + j] = hb;
      if (writeSeq) hseq[((size_t)(bbase + m) * SEQ + t) * HIDDEN + hidbase + j] = hb;
    }
    group_barrier(ctr, gen);
    // gather the full h_t (all 512 hidden) for our 16 batch rows
    {
      const uint4* S = (const uint4*)(hstate + (size_t)bbase * HIDDEN);
      uint4* D = (uint4*)u;
      for (int i = tid; i < 16 * HIDDEN / 8; i += 256) D[i] = S[i];
    }
    __syncthreads();
  }
}

// ---------------------------------------------------------------------------
// Phase 3: out = h_last @ W_fc^T + b_fc. One 16x16 WMMA tile per 32-thr block.
// ---------------------------------------------------------------------------
__global__ void __launch_bounds__(32) k_fc(const u16* __restrict__ h,
    const u16* __restrict__ Wfc, const float* __restrict__ bfc,
    float* __restrict__ out)
{
  const int mt = blockIdx.x, nt = blockIdx.y;
  const int lane = threadIdx.x & 31, half = lane >> 4, l16 = lane & 15;
  f32x8 acc;
  float bv = bfc[nt * 16 + l16];
#pragma unroll
  for (int v = 0; v < 8; ++v) acc[v] = bv;
  const u16* Arow = h + (size_t)(mt * 16) * HIDDEN;
  const u16* Brow = Wfc + (size_t)(nt * 16) * HIDDEN;
#pragma unroll
  for (int kt = 0; kt < HIDDEN / 32; ++kt) {
    bf16x16 a = load_a(Arow, HIDDEN, kt * 32, l16, half);
    bf16x16 b = load_b(Brow, HIDDEN, kt * 32, l16, half);
    acc = wmma_bf16(a, b, acc);
  }
#pragma unroll
  for (int v = 0; v < 8; ++v)
    out[(size_t)(mt * 16 + v + half * 8) * NKEYS + nt * 16 + l16] = acc[v];
}

// ---------------------------------------------------------------------------
extern "C" void kernel_launch(void* const* d_in, const int* in_sizes, int n_in,
                              void* d_out, int out_size, void* d_ws, size_t ws_size,
                              hipStream_t stream)
{
  const float* x     = (const float*)d_in[0];
  const float* Wih0f = (const float*)d_in[1];
  const float* Whh0f = (const float*)d_in[2];
  const float* bih0  = (const float*)d_in[3];
  const float* bhh0  = (const float*)d_in[4];
  const float* Wih1f = (const float*)d_in[5];
  const float* Whh1f = (const float*)d_in[6];
  const float* bih1  = (const float*)d_in[7];
  const float* bhh1  = (const float*)d_in[8];
  const float* Wfcf  = (const float*)d_in[9];
  const float* bfc   = (const float*)d_in[10];

  char* ws = (char*)d_ws;
  u32*   BAR   = (u32*)(ws + BAR_OFF);
  float* BIAS0 = (float*)(ws + BIAS0_OFF);
  float* BIAS1 = (float*)(ws + BIAS1_OFF);
  u16*   WIH0  = (u16*)(ws + WIH0_OFF);
  u16*   WHH0  = (u16*)(ws + WHH0_OFF);
  u16*   WIH1  = (u16*)(ws + WIH1_OFF);
  u16*   WHH1  = (u16*)(ws + WHH1_OFF);
  u16*   WFC   = (u16*)(ws + WFC_OFF);
  u16*   HST   = (u16*)(ws + HST_OFF);
  u16*   HSEQ  = (u16*)(ws + HSEQ_OFF);
  float* XG    = (float*)(ws + XG_OFF);

  // opt in to >64KB dynamic LDS (CDNA5: 320KB per workgroup)
  (void)hipFuncSetAttribute((const void*)k_xg_gemm<128, true>,
                            hipFuncAttributeMaxDynamicSharedMemorySize, 2 * 128 * 128 * 2);
  (void)hipFuncSetAttribute((const void*)k_xg_gemm<512, false>,
                            hipFuncAttributeMaxDynamicSharedMemorySize, 2 * 128 * 512 * 2);
  (void)hipFuncSetAttribute((const void*)k_lstm,
                            hipFuncAttributeMaxDynamicSharedMemorySize, LSTM_SMEM);

  // prep: barrier state, bf16 weights, fused biases
  k_zero<<<1, 64, 0, stream>>>(BAR, 64);
  k_f2b<<<(GATES * NKEYS  + 255) / 256, 256, 0, stream>>>(WIH0, Wih0f, GATES * NKEYS);
  k_f2b<<<(GATES * HIDDEN + 255) / 256, 256, 0, stream>>>(WHH0, Whh0f, GATES * HIDDEN);
  k_f2b<<<(GATES * HIDDEN + 255) / 256, 256, 0, stream>>>(WIH1, Wih1f, GATES * HIDDEN);
  k_f2b<<<(GATES * HIDDEN + 255) / 256, 256, 0, stream>>>(WHH1, Whh1f, GATES * HIDDEN);
  k_f2b<<<(NKEYS * HIDDEN + 255) / 256, 256, 0, stream>>>(WFC, Wfcf, NKEYS * HIDDEN);
  k_bias<<<(GATES + 255) / 256, 256, 0, stream>>>(BIAS0, bih0, bhh0, GATES);
  k_bias<<<(GATES + 255) / 256, 256, 0, stream>>>(BIAS1, bih1, bhh1, GATES);

  // layer 0
  k_xg_gemm<128, true><<<dim3(16, MROWS / 128), 256, 2 * 128 * 128 * 2, stream>>>(
      XG, (const void*)x, WIH0, BIAS0);
  k_lstm<<<256, 256, LSTM_SMEM, stream>>>(XG, WHH0, HST, HSEQ, BAR, 1);

  // layer 1 (xg buffer reused)
  k_xg_gemm<512, false><<<dim3(16, MROWS / 128), 256, 2 * 128 * 512 * 2, stream>>>(
      XG, (const void*)HSEQ, WIH1, BIAS1);
  k_lstm<<<256, 256, LSTM_SMEM, stream>>>(XG, WHH1, HST, HSEQ, BAR, 0);

  // head
  k_fc<<<dim3(16, 8), 32, 0, stream>>>(HST, WFC, bfc, (float*)d_out);

  (void)in_sizes; (void)n_in; (void)out_size; (void)ws_size;
}
